// MultiQueryAttention_72524817760462
// MI455X (gfx1250) — matrix operand order repaired and checked by reading, not compiled, not run on hardware
//
#include <hip/hip_runtime.h>
#include <hip/hip_bf16.h>
#include <stdint.h>

// ---------------- problem constants (match reference) ----------------
constexpr int Bb  = 2;
constexpr int Ss  = 2048;
constexpr int Hh  = 2048;
constexpr int NHh = 16;
constexpr int HDd = 128;   // head_dim
constexpr int KT  = 32;    // key-tile (contraction tile for P.V)

#define DEVINL __device__ __forceinline__

typedef __attribute__((ext_vector_type(16))) __bf16 bf16x16;
typedef __attribute__((ext_vector_type(8)))  float  f32x8;
typedef __attribute__((ext_vector_type(4)))  unsigned int u32x4;
typedef __attribute__((ext_vector_type(8)))  int          i32x8;
typedef __attribute__((ext_vector_type(4)))  int          i32x4;

struct U4x2 { uint4 lo, hi; };

DEVINL bf16x16 make_frag(uint4 lo, uint4 hi) {
  U4x2 t{lo, hi};
  return __builtin_bit_cast(bf16x16, t);
}

DEVINL uint16_t f32_to_bf16_bits(float f) {
  uint32_t u = __builtin_bit_cast(uint32_t, f);
  u += 0x7fffu + ((u >> 16) & 1u);   // round-to-nearest-even
  return (uint16_t)(u >> 16);
}

DEVINL f32x8 wmma_bf16(bf16x16 a, bf16x16 b, f32x8 c) {
  // D = A(16x32 bf16) * B(32x16 bf16) + C(16x16 f32)
  return __builtin_amdgcn_wmma_f32_16x16x32_bf16(
      false, a, false, b, (short)0, c, false, false);
}

// ---------------- TDM: 2D tile DMA global -> LDS (bf16 elements) ----------------
// Builds the D# per CDNA5 ISA 8.3/8.4 and issues TENSOR_LOAD_TO_LDS.
// This toolchain exposes the 6-arg builtin:
//   (u32x4 group0, i32x8 group1, i32x4 group2, i32x4 group3, i32x8 extra, i32 cpol)
// lds_off     : LDS byte address of tile destination (single __shared__ symbol @0)
// gaddr       : global byte address of tile start
// tensor_d0/1 : tensor extents (elements) for OOB
// tile_d0/1   : tile extents (elements); rows stored compacted in LDS
// stride0     : elements between consecutive dim-1 lines
DEVINL void tdm_load_2d(uint32_t lds_off, const void* gaddr,
                        uint32_t tensor_d0, uint32_t tensor_d1,
                        uint32_t tile_d0, uint32_t tile_d1, uint64_t stride0) {
  const uint64_t ga = (uint64_t)(uintptr_t)gaddr;
  u32x4 g0;
  g0[0] = 1u;                                               // count=1 (valid user D#)
  g0[1] = lds_off;                                          // lds_addr (bytes)
  g0[2] = (uint32_t)ga;                                     // global_addr[31:0]
  g0[3] = (uint32_t)((ga >> 32) & 0x01FFFFFFu) | (2u << 30); // ga[56:32] | type=2
  i32x8 g1;
  g1[0] = (int)(1u << 16);                                  // data_size=1 (2 bytes)
  g1[1] = (int)((tensor_d0 & 0xFFFFu) << 16);               // tensor_dim0[15:0]
  g1[2] = (int)(((tensor_d0 >> 16) & 0xFFFFu) |
                ((tensor_d1 & 0xFFFFu) << 16));             // td0[31:16] | td1[15:0]
  g1[3] = (int)(((tensor_d1 >> 16) & 0xFFFFu) |
                ((tile_d0 & 0xFFFFu) << 16));               // td1[31:16] | tile_dim0
  g1[4] = (int)(tile_d1 & 0xFFFFu);                         // tile_dim1 (tile_dim2=0)
  g1[5] = (int)(uint32_t)(stride0 & 0xFFFFFFFFu);           // tensor_dim0_stride lo
  g1[6] = (int)(uint32_t)((stride0 >> 32) & 0xFFFFu);       // stride hi (dim1_stride=0)
  g1[7] = 0;
  i32x4 z4 = {};
  i32x8 z8 = {};
  __builtin_amdgcn_tensor_load_to_lds(g0, g1, z4, z4, z8, 0);
}

// ---------------- fp32 -> bf16 convert ----------------
__global__ void cvt_f32_bf16(const float* __restrict__ src,
                             uint16_t* __restrict__ dst, int n) {
  int i = blockIdx.x * blockDim.x + threadIdx.x;
  int stride = gridDim.x * blockDim.x;
  for (; i < n; i += stride) dst[i] = f32_to_bf16_bits(src[i]);
}

// ---------------- generic WMMA GEMM: out = A[M,K] * W[N,K]^T + bias ----------------
// 64x32 tile per wave: 4 A-frags x 2 B-frags -> 8 WMMAs per 32-K step.
// mode 0: fp32 out, row-major [M,N]                     (O projection -> d_out)
// mode 1: bf16 out, q layout ((b*NH+nh)*S + s)*HD + hd  (Q projection)
// mode 2: bf16 out, row-major [M,N]                     (K projection)
// mode 3: bf16 out, transposed per batch (b*HD + n)*S+s (V projection -> V^T)
__global__ __launch_bounds__(256)
void gemm_bf16_wmma(const uint16_t* __restrict__ A, const uint16_t* __restrict__ W,
                    const float* __restrict__ bias, void* __restrict__ outp,
                    int M, int N, int K, int mode) {
  const int lane = threadIdx.x & 31;
  const int lrow = lane & 15;
  const int half = lane >> 4;
  const int wave = blockIdx.x * (blockDim.x >> 5) + (threadIdx.x >> 5);
  const int ntn  = N >> 5;
  const int tm = wave / ntn;
  const int tn = wave % ntn;
  if (tm * 64 >= M) return;

  const uint16_t* arow[4];
#pragma unroll
  for (int i = 0; i < 4; ++i)
    arow[i] = A + (size_t)(tm * 64 + i * 16 + lrow) * K;
  const uint16_t* wrow[2];
#pragma unroll
  for (int j = 0; j < 2; ++j)
    wrow[j] = W + (size_t)(tn * 32 + j * 16 + lrow) * K;

  const int aoff0 = half * 8;            // A-frag interleaved chunks
  const int aoff1 = 16 + half * 8;
  const int boff0 = half * 8;            // B-frag interleaved chunks
  const int boff1 = 16 + half * 8;

  f32x8 acc[4][2] = {};
  for (int k0 = 0; k0 < K; k0 += 32) {
    bf16x16 bfr[2];
#pragma unroll
    for (int j = 0; j < 2; ++j)
      bfr[j] = make_frag(*(const uint4*)(wrow[j] + k0 + boff0),
                         *(const uint4*)(wrow[j] + k0 + boff1));
#pragma unroll
    for (int i = 0; i < 4; ++i) {
      bf16x16 af = make_frag(*(const uint4*)(arow[i] + k0 + aoff0),
                             *(const uint4*)(arow[i] + k0 + aoff1));
      acc[i][0] = wmma_bf16(af, bfr[0], acc[i][0]);
      acc[i][1] = wmma_bf16(af, bfr[1], acc[i][1]);
    }
  }

  float*    outF = (float*)outp;
  uint16_t* outB = (uint16_t*)outp;
#pragma unroll
  for (int i = 0; i < 4; ++i) {
#pragma unroll
    for (int j = 0; j < 2; ++j) {
      const int col = tn * 32 + j * 16 + lrow;
      const float bvv = bias ? bias[col] : 0.0f;
#pragma unroll
      for (int r = 0; r < 8; ++r) {
        const int row = tm * 64 + i * 16 + r + half * 8;
        const float v = acc[i][j][r] + bvv;
        if (mode == 0) {
          outF[(size_t)row * N + col] = v;
        } else if (mode == 1) {
          const int bb = row / Ss, s = row % Ss;
          const int nh = col / HDd, hd = col % HDd;
          outB[(((size_t)(bb * NHh + nh)) * Ss + s) * HDd + hd] = f32_to_bf16_bits(v);
        } else if (mode == 2) {
          outB[(size_t)row * N + col] = f32_to_bf16_bits(v);
        } else { // mode 3: V^T
          const int bb = row / Ss, s = row % Ss;
          outB[((size_t)(bb * HDd + col)) * Ss + s] = f32_to_bf16_bits(v);
        }
      }
    }
  }
}

// ---------------- flash-style MQA attention with TDM-staged K/V ----------------
// grid: B*NH*(S/128) blocks, 256 threads (8 waves); wave owns 16 query rows.
// K tile (32x128, contiguous 8KB) and V^T tile (128x32, stride S) are DMA'd into
// LDS by the Tensor Data Mover, double-buffered, shared by all 8 waves.
// LDS map (uint16 elems): K0 @0, K1 @4096, V0 @8192, V1 @12288, P @16384+w*512
__global__ __launch_bounds__(256)
void mqa_attn(const uint16_t* __restrict__ qb, const uint16_t* __restrict__ kb,
              const uint16_t* __restrict__ vtb, uint16_t* __restrict__ ab) {
  __shared__ __align__(16) uint16_t smem[4 * KT * HDd + 8 * 16 * KT];
  constexpr uint32_t KOFF[2] = {0u, (uint32_t)(KT * HDd)};
  constexpr uint32_t VOFF[2] = {(uint32_t)(2 * KT * HDd), (uint32_t)(3 * KT * HDd)};
  constexpr uint32_t POFF    = (uint32_t)(4 * KT * HDd);

  const int lane = threadIdx.x & 31;
  const int lrow = lane & 15;
  const int half = lane >> 4;
  const int w    = threadIdx.x >> 5;

  const int qtiles = Ss / 128;
  const int qt = blockIdx.x % qtiles;
  const int bh = blockIdx.x / qtiles;          // b*NH + nh
  const int b  = bh / NHh;
  const int nh = bh % NHh;
  const int qbase = qt * 128 + w * 16;

  const uint16_t* Q  = qb  + (size_t)bh * Ss * HDd;
  const uint16_t* Kp = kb  + (size_t)b  * Ss * HDd;
  const uint16_t* Vt = vtb + (size_t)b  * HDd * Ss;

  // Kick off TDM for tile 0 (wave 0 only; TENSORcnt is per-wave).
  if (w == 0) {
    // K tile: 32 rows x HD=128 -> fully contiguous, 1D copy of 4096 elems
    tdm_load_2d(KOFF[0] * 2, Kp, KT * HDd, 1, KT * HDd, 1, KT * HDd);
    // V^T tile: 128 rows x 32 cols, row stride S
    tdm_load_2d(VOFF[0] * 2, Vt, Ss, HDd, KT, HDd, Ss);
  }

  // Preload Q fragments for all 4 contraction steps over HD=128
  bf16x16 qf[4];
  {
    const uint16_t* qrow = Q + (size_t)(qbase + lrow) * HDd;
#pragma unroll
    for (int d = 0; d < 4; ++d)
      qf[d] = make_frag(*(const uint4*)(qrow + d * 32 + half * 8),
                        *(const uint4*)(qrow + d * 32 + 16 + half * 8));
  }

  f32x8 oacc[8] = {};                      // 16 rows x 128 cols fp32
  float mrun[8], lrun[8];
#pragma unroll
  for (int r = 0; r < 8; ++r) { mrun[r] = -1e30f; lrun[r] = 0.0f; }
  const float scl = 0.08838834764831845f;  // 1/sqrt(128)

  const int NT = Ss / KT;
  for (int it = 0; it < NT; ++it) {
    const int buf = it & 1;
    if (w == 0) __builtin_amdgcn_s_wait_tensorcnt(0);  // tile `it` landed in LDS
    __syncthreads();                                   // release all waves
    if (w == 0 && it + 1 < NT) {                       // prefetch tile it+1
      const int t1 = (it + 1) * KT;
      tdm_load_2d(KOFF[buf ^ 1] * 2, Kp + (size_t)t1 * HDd,
                  KT * HDd, 1, KT * HDd, 1, KT * HDd);
      tdm_load_2d(VOFF[buf ^ 1] * 2, Vt + t1, Ss, HDd, KT, HDd, Ss);
    }

    const uint16_t* Kl = &smem[KOFF[buf]];   // [key][128]
    const uint16_t* Vl = &smem[VOFF[buf]];   // [dcol][32]

    // ---- scores: S[16 x 32] = Q(16xHD) . K^T, K from LDS ----
    f32x8 sc[2] = {};
#pragma unroll
    for (int nt = 0; nt < 2; ++nt) {
      const uint16_t* krow = Kl + (nt * 16 + lrow) * HDd;
#pragma unroll
      for (int d = 0; d < 4; ++d) {
        bf16x16 kf = make_frag(*(const uint4*)(krow + d * 32 + half * 8),
                               *(const uint4*)(krow + d * 32 + 16 + half * 8));
        sc[nt] = wmma_bf16(qf[d], kf, sc[nt]);
      }
    }

    // ---- online softmax (rows live in 16-lane halves of C/D layout) ----
    uint16_t* pb = &smem[POFF + w * (16 * KT)];
    float corr[8];
#pragma unroll
    for (int r = 0; r < 8; ++r) {
      float s0 = sc[0][r] * scl;
      float s1 = sc[1][r] * scl;
      float mx = fmaxf(s0, s1);
      mx = fmaxf(mx, __shfl_xor(mx, 1, 32));
      mx = fmaxf(mx, __shfl_xor(mx, 2, 32));
      mx = fmaxf(mx, __shfl_xor(mx, 4, 32));
      mx = fmaxf(mx, __shfl_xor(mx, 8, 32));
      const float mnew = fmaxf(mrun[r], mx);
      const float c  = __expf(mrun[r] - mnew);
      const float p0 = __expf(s0 - mnew);
      const float p1 = __expf(s1 - mnew);
      float ps = p0 + p1;
      ps += __shfl_xor(ps, 1, 32);
      ps += __shfl_xor(ps, 2, 32);
      ps += __shfl_xor(ps, 4, 32);
      ps += __shfl_xor(ps, 8, 32);
      lrun[r] = lrun[r] * c + ps;
      mrun[r] = mnew;
      corr[r] = c;
      const int prow = r + half * 8;
      pb[prow * KT + lrow]      = f32_to_bf16_bits(p0);
      pb[prow * KT + 16 + lrow] = f32_to_bf16_bits(p1);
    }
#pragma unroll
    for (int nt = 0; nt < 8; ++nt)
#pragma unroll
      for (int r = 0; r < 8; ++r)
        oacc[nt][r] *= corr[r];

    // Reload P in A-fragment layout (same-wave LDS RAW: DS unit is in-order).
    bf16x16 pf;
    {
      const uint16_t* prow = pb + lrow * KT;
      pf = make_frag(*(const uint4*)(prow + half * 8),
                     *(const uint4*)(prow + 16 + half * 8));
    }

    // ---- O += P(16x32) . V(32x128), V columns from LDS ----
#pragma unroll
    for (int nt = 0; nt < 8; ++nt) {
      const uint16_t* vrow = Vl + (nt * 16 + lrow) * KT;
      bf16x16 vf = make_frag(*(const uint4*)(vrow + half * 8),
                             *(const uint4*)(vrow + 16 + half * 8));
      oacc[nt] = wmma_bf16(pf, vf, oacc[nt]);
    }
    __syncthreads();   // all waves done with this K/V buffer before it is refilled
  }

  // ---- normalize and write merged-head output [B, S, H] ----
#pragma unroll
  for (int nt = 0; nt < 8; ++nt) {
    const int col = nh * HDd + nt * 16 + lrow;
#pragma unroll
    for (int r = 0; r < 8; ++r) {
      const int row = qbase + r + half * 8;
      const float o = oacc[nt][r] / lrun[r];
      ab[((size_t)(b * Ss + row)) * Hh + col] = f32_to_bf16_bits(o);
    }
  }
}

// ---------------- host launcher ----------------
extern "C" void kernel_launch(void* const* d_in, const int* in_sizes, int n_in,
                              void* d_out, int out_size, void* d_ws, size_t ws_size,
                              hipStream_t stream) {
  (void)in_sizes; (void)n_in; (void)out_size; (void)ws_size;

  const float* x  = (const float*)d_in[0];
  const float* Wq = (const float*)d_in[1];
  const float* bq = (const float*)d_in[2];
  const float* Wk = (const float*)d_in[3];
  const float* bk = (const float*)d_in[4];
  const float* Wv = (const float*)d_in[5];
  const float* bv = (const float*)d_in[6];
  const float* Wo = (const float*)d_in[7];
  const float* bo = (const float*)d_in[8];
  float* out = (float*)d_out;

  uint16_t* ws = (uint16_t*)d_ws;
  size_t o = 0;
  uint16_t* xb  = ws + o;  o += (size_t)Bb * Ss * Hh;     // x bf16
  uint16_t* wqb = ws + o;  o += (size_t)Hh * Hh;          // Wq bf16
  uint16_t* wkb = ws + o;  o += (size_t)HDd * Hh;         // Wk bf16
  uint16_t* wvb = ws + o;  o += (size_t)HDd * Hh;         // Wv bf16
  uint16_t* wob = ws + o;  o += (size_t)Hh * Hh;          // Wo bf16
  uint16_t* qbv = ws + o;  o += (size_t)Bb * Ss * Hh;     // Q  [B*NH, S, HD]
  uint16_t* kbv = ws + o;  o += (size_t)Bb * Ss * HDd;    // K  [B, S, HD]
  uint16_t* vtb = ws + o;  o += (size_t)Bb * HDd * Ss;    // V^T [B, HD, S]
  uint16_t* abv = ws + o;  o += (size_t)Bb * Ss * Hh;     // attn out [B, S, H]

  auto cvt = [&](const float* s, uint16_t* d, int n) {
    int blocks = (n + 256 * 8 - 1) / (256 * 8);
    cvt_f32_bf16<<<blocks, 256, 0, stream>>>(s, d, n);
  };
  cvt(x,  xb,  Bb * Ss * Hh);
  cvt(Wq, wqb, Hh * Hh);
  cvt(Wk, wkb, HDd * Hh);
  cvt(Wv, wvb, HDd * Hh);
  cvt(Wo, wob, Hh * Hh);

  const int M = Bb * Ss;  // 4096
  // Q projection -> head-major layout   (64x32 tiles, 8 waves/block)
  gemm_bf16_wmma<<<(M / 64) * (Hh / 32) / 8, 256, 0, stream>>>(
      xb, wqb, bq, qbv, M, Hh, Hh, /*mode=*/1);
  // K projection -> [B,S,HD]
  gemm_bf16_wmma<<<(M / 64) * (HDd / 32) / 8, 256, 0, stream>>>(
      xb, wkb, bk, kbv, M, HDd, Hh, /*mode=*/2);
  // V projection -> transposed [B,HD,S]
  gemm_bf16_wmma<<<(M / 64) * (HDd / 32) / 8, 256, 0, stream>>>(
      xb, wvb, bv, vtb, M, HDd, Hh, /*mode=*/3);
  // attention (TDM-staged K/V)
  mqa_attn<<<Bb * NHh * (Ss / 128), 256, 0, stream>>>(qbv, kbv, vtb, abv);
  // O projection -> fp32 d_out
  gemm_bf16_wmma<<<(M / 64) * (Hh / 32) / 8, 256, 0, stream>>>(
      abv, wob, bo, out, M, Hh, Hh, /*mode=*/0);
}
